// Head_10642928959521
// MI455X (gfx1250) — matrix operand order repaired and checked by reading, not compiled, not run on hardware
//
#include <hip/hip_runtime.h>

// ---------------------------------------------------------------------------
// Causal attention head (B=8, T=4096, C=384, H=64) for gfx1250 (MI455X).
// Kernel 0: one-shot W -> f16 transpose [3][H][C]; C^-0.5 * log2(e) folded
//           into Wq so attention softmax runs in native base-2 (v_exp_f32).
// Kernel 1: Q/K/V projections via v_wmma_f32_16x16x32_f16; Q,K row-major f16,
//           V transposed f16 (Vt[h][B*T]) for contiguous PV B-fragments.
// Kernel 2: flash-attention forward, 16 query rows per wave32, 64-key blocks,
//           online base-2 softmax, P transposed D->A via per-wave LDS staging.
// ---------------------------------------------------------------------------

typedef _Float16 v16h __attribute__((ext_vector_type(16)));
typedef _Float16 v8h  __attribute__((ext_vector_type(8)));
typedef float    v8f  __attribute__((ext_vector_type(8)));

constexpr int Bn = 8, Tn = 4096, Cn = 384, Hn = 64;
constexpr size_t BT = (size_t)Bn * Tn;

static __device__ __forceinline__ v8f wmma_f16(v16h a, v16h b, v8f c) {
  // 8 args: (neg_a, A, neg_b, B, c_mod, C, reuse_a, reuse_b)
  return __builtin_amdgcn_wmma_f32_16x16x32_f16(false, a, false, b,
                                                (short)0, c, false, false);
}

// ---------------------------------------------------------------------------
// Kernel 0: Wt[mat][h][k] = (f16) W[mat][k][h] * (mat==Q ? C^-0.5*log2e : 1)
// One-time layout conversion so the projection B-fragments are contiguous.
// ---------------------------------------------------------------------------
__global__ __launch_bounds__(256) void prep_w(
    const float* __restrict__ Wq, const float* __restrict__ Wk,
    const float* __restrict__ Wv, _Float16* __restrict__ Wt)
{
  int id = blockIdx.x * blockDim.x + threadIdx.x;
  if (id >= 3 * Cn * Hn) return;
  int mat = id / (Cn * Hn);
  int rem = id % (Cn * Hn);
  int h   = rem / Cn;
  int k   = rem % Cn;
  const float* W = (mat == 0) ? Wq : (mat == 1) ? Wk : Wv;
  // base-2 softmax: logits pre-scaled by log2(e) so p = exp2(s - m)
  float scale = (mat == 0) ? rsqrtf((float)Cn) * 1.44269504088896340736f : 1.0f;
  Wt[(size_t)mat * Cn * Hn + (size_t)h * Cn + k] =
      (_Float16)(W[(size_t)k * Hn + h] * scale);
}

// ---------------------------------------------------------------------------
// Kernel 1: [B*T, C] x [C, H] -> Q,K (row-major f16) and Vt (transposed f16).
// One wave per 16-row tile; 4 col tiles x 3 matrices accumulators.
// ---------------------------------------------------------------------------
__global__ __launch_bounds__(128) void proj_qkv(
    const float* __restrict__ x, const _Float16* __restrict__ Wt,
    _Float16* __restrict__ Qh, _Float16* __restrict__ Kh,
    _Float16* __restrict__ Vt)
{
  const int lane  = threadIdx.x & 31;
  const int gwave = (blockIdx.x * blockDim.x + threadIdx.x) >> 5;
  const int row0  = gwave * 16;          // tile of 16 rows in flattened [B*T)
  const int n     = lane & 15;
  const int g     = lane >> 4;

  v8f acc[3][4] = {};                    // {q,k,v} x 4 column tiles of 16

  for (int k0 = 0; k0 < Cn; k0 += 32) {
    // A fragment: row m = lane&15, element i -> K = 16g + i
    const float* xrow = x + (size_t)(row0 + n) * Cn + k0 + g * 16;
    v16h a;
    #pragma unroll
    for (int i = 0; i < 8; ++i) {
      a[i]     = (_Float16)xrow[i];
      a[i + 8] = (_Float16)xrow[8 + i];
    }
    #pragma unroll
    for (int mat = 0; mat < 3; ++mat) {
      #pragma unroll
      for (int ct = 0; ct < 4; ++ct) {
        // B fragment: col n -> h = ct*16+n, element i -> k = k0 + 16g + i
        // Wt[mat][h][k]: contiguous 16 halves per lane (32B aligned).
        v16h bf = *(const v16h*)(Wt + ((size_t)mat * Hn + ct * 16 + n) * Cn +
                                 k0 + g * 16);
        acc[mat][ct] = wmma_f16(a, bf, acc[mat][ct]);
      }
    }
  }

  #pragma unroll
  for (int ct = 0; ct < 4; ++ct) {
    #pragma unroll
    for (int r = 0; r < 8; ++r) {
      // D layout: vgpr r -> row M = r + 8g, col N = n
      size_t idx = (size_t)(row0 + r + 8 * g) * Hn + ct * 16 + n;
      Qh[idx] = (_Float16)acc[0][ct][r];   // scale already folded into Wq
      Kh[idx] = (_Float16)acc[1][ct][r];
    }
    // V transposed: Vt[h][row]; rows r+8g are contiguous -> one 16B store
    v8h vv;
    #pragma unroll
    for (int r = 0; r < 8; ++r) vv[r] = (_Float16)acc[2][ct][r];
    *(v8h*)(Vt + (size_t)(ct * 16 + n) * BT + row0 + 8 * g) = vv;
  }
}

// ---------------------------------------------------------------------------
// Kernel 2: flash attention forward. One wave per (b, 16-row query tile).
// Key blocks of 64: 8 WMMA for S, one softmax pass, 8 WMMA for O += P*V.
// ---------------------------------------------------------------------------
__global__ __launch_bounds__(128) void attn_fwd(
    const _Float16* __restrict__ Qh, const _Float16* __restrict__ Kh,
    const _Float16* __restrict__ Vt, float* __restrict__ out)
{
  __shared__ _Float16 lds_p[4][16][64];    // per-wave P staging (D->A layout)

  const int lane  = threadIdx.x & 31;
  const int wid   = threadIdx.x >> 5;
  const int gwave = blockIdx.x * (blockDim.x >> 5) + wid;
  const int b     = gwave / (Tn / 16);
  const int qt    = gwave % (Tn / 16);
  const int qrow0 = qt * 16;               // local (per-batch) first query row
  const size_t rb = (size_t)b * Tn;        // flattened row base for this batch
  const int n     = lane & 15;
  const int g     = lane >> 4;

  // Q A-fragments, two 32-wide K chunks of the head dim (H = 64)
  v16h aq[2];
  {
    const _Float16* qrow = Qh + (rb + qrow0 + n) * (size_t)Hn;
    #pragma unroll
    for (int c = 0; c < 2; ++c) {
      v8h lo = *(const v8h*)(qrow + c * 32 + g * 16);
      v8h hi = *(const v8h*)(qrow + c * 32 + 8 + g * 16);
      #pragma unroll
      for (int i = 0; i < 8; ++i) { aq[c][i] = lo[i]; aq[c][i + 8] = hi[i]; }
    }
  }

  v8f o[4] = {};                           // O accumulators, 4 col tiles
  float mrow[8], lrow[8];                  // per-row running max / sum (base-2)
  #pragma unroll
  for (int r = 0; r < 8; ++r) { mrow[r] = -__builtin_inff(); lrow[r] = 0.f; }

  const int qlast = qrow0 + 15;
  for (int j0 = 0; j0 <= qlast; j0 += 64) {
    // ---- S = Q * K^T for keys [j0, j0+64): four 16x16 tiles ------------
    v8f s[4] = {};
    #pragma unroll
    for (int half = 0; half < 4; ++half) {
      // B fragment: col n = key-in-tile, element i -> h = 32c + 16g + i
      const _Float16* krow =
          Kh + (rb + j0 + half * 16 + n) * (size_t)Hn + g * 16;
      #pragma unroll
      for (int c = 0; c < 2; ++c) {
        v16h bk = *(const v16h*)(krow + c * 32);   // contiguous 32 B
        s[half] = wmma_f16(aq[c], bk, s[half]);
      }
    }

    // ---- causal mask (only blocks that straddle the diagonal) ----------
    if (j0 + 63 > qrow0) {
      #pragma unroll
      for (int half = 0; half < 4; ++half)
        #pragma unroll
        for (int r = 0; r < 8; ++r)
          if (j0 + half * 16 + n > qrow0 + r + 8 * g)
            s[half][r] = -__builtin_inff();
    }

    // ---- online softmax (base 2): rows across 16 lanes sharing g -------
    #pragma unroll
    for (int r = 0; r < 8; ++r) {
      float v = fmaxf(fmaxf(s[0][r], s[1][r]), fmaxf(s[2][r], s[3][r]));
      #pragma unroll
      for (int off = 1; off < 16; off <<= 1)
        v = fmaxf(v, __shfl_xor(v, off, 16));
      float mnew = fmaxf(mrow[r], v);
      float p[4];
      #pragma unroll
      for (int half = 0; half < 4; ++half)
        p[half] = exp2f(s[half][r] - mnew);
      float alpha = exp2f(mrow[r] - mnew);
      mrow[r] = mnew;
      float ps = (p[0] + p[1]) + (p[2] + p[3]);
      #pragma unroll
      for (int off = 1; off < 16; off <<= 1)
        ps += __shfl_xor(ps, off, 16);
      lrow[r] = lrow[r] * alpha + ps;
      #pragma unroll
      for (int ct = 0; ct < 4; ++ct) o[ct][r] *= alpha;
      // stash P in row-major LDS for the D->A layout transpose
      #pragma unroll
      for (int half = 0; half < 4; ++half)
        lds_p[wid][r + 8 * g][half * 16 + n] = (_Float16)p[half];
    }

    // per-wave DS stores -> loads: wait on DScnt (same wave, in order)
    asm volatile("s_wait_dscnt 0" ::: "memory");

    // ---- P as two A fragments (16x32 each, K = keys) -------------------
    v16h ap[2];
    #pragma unroll
    for (int kc = 0; kc < 2; ++kc) {
      const _Float16* prow = &lds_p[wid][n][kc * 32];
      v8h lo = *(const v8h*)(prow + g * 16);
      v8h hi = *(const v8h*)(prow + 8 + g * 16);
      #pragma unroll
      for (int i = 0; i < 8; ++i) { ap[kc][i] = lo[i]; ap[kc][i + 8] = hi[i]; }
    }

    // ---- O += P * V (Vt is transposed: key index contiguous) -----------
    #pragma unroll
    for (int kc = 0; kc < 2; ++kc) {
      #pragma unroll
      for (int ct = 0; ct < 4; ++ct) {
        // B fragment: col n -> h = ct*16+n, element i -> key = j0+32kc+16g+i
        v16h bv = *(const v16h*)(Vt + (size_t)(ct * 16 + n) * BT + rb + j0 +
                                 kc * 32 + g * 16);
        o[ct] = wmma_f16(ap[kc], bv, o[ct]);
      }
    }
  }

  // ---- epilogue: normalize and store (f32 output) ----------------------
  #pragma unroll
  for (int ct = 0; ct < 4; ++ct)
    #pragma unroll
    for (int r = 0; r < 8; ++r)
      out[(rb + qrow0 + r + 8 * g) * (size_t)Hn + ct * 16 + n] =
          o[ct][r] / lrow[r];
}

// ---------------------------------------------------------------------------
extern "C" void kernel_launch(void* const* d_in, const int* in_sizes, int n_in,
                              void* d_out, int out_size, void* d_ws,
                              size_t ws_size, hipStream_t stream) {
  const float* x  = (const float*)d_in[0];
  const float* Wk = (const float*)d_in[1];
  const float* Wq = (const float*)d_in[2];
  const float* Wv = (const float*)d_in[3];
  float* out = (float*)d_out;

  // Workspace: Q, K (row-major), Vt (transposed) f16 + Wt f16
  _Float16* Qh = (_Float16*)d_ws;          // BT*Hn halves (4 MB)
  _Float16* Kh = Qh + BT * Hn;             // 4 MB
  _Float16* Vt = Kh + BT * Hn;             // 4 MB
  _Float16* Wt = Vt + BT * Hn;             // 3*Cn*Hn halves (144 KB)

  constexpr int waves = (Bn * Tn) / 16;     // 2048 waves, 16 rows each
  constexpr int wavesPerBlock = 4;          // 128 threads (wave32)
  dim3 grid(waves / wavesPerBlock), block(32 * wavesPerBlock);

  prep_w<<<(3 * Cn * Hn + 255) / 256, 256, 0, stream>>>(Wq, Wk, Wv, Wt);
  proj_qkv<<<grid, block, 0, stream>>>(x, Wt, Qh, Kh, Vt);
  attn_fwd<<<grid, block, 0, stream>>>(Qh, Kh, Vt, out);
}
